// CustomModel_16681652978184
// MI455X (gfx1250) — matrix operand order repaired and checked by reading, not compiled, not run on hardware
//
#include <hip/hip_runtime.h>
#include <stdint.h>

// ---------------------------------------------------------------------------
// 3-layer LSTM (T=4096, B=6) for MI455X / gfx1250.
//
//  * bf16 WMMA 16x16x32 with f32 accumulation; batch 6 padded to M=16
//    (padding rows clamp to row 0; their results are never stored).
//  * Per layer: persistent kernel, H/16 workgroups x 128 threads (4 waves).
//    Wave g computes gate-g's 16x16 z-tile for a 16-unit slice. Weights are
//    pre-packed to the ISA B-fragment lane layout and parked in LDS
//    (<=128KB/WG of 320KB) for all 4096 steps via async global->LDS copies,
//    so the recurrence moves only h (6KB) through L2 per step.
//  * c-state lives in VGPRs across the whole T loop.
//  * Grid-wide release/acquire barrier per timestep (atomic counter+s_sleep).
//  * Layer2 inherits layer1 final (h,c); layer3 starts fresh (ref semantics).
// ---------------------------------------------------------------------------

typedef unsigned short u16;
typedef __attribute__((ext_vector_type(16))) __bf16 v16bf;
typedef __attribute__((ext_vector_type(8)))  float  v8f;

#define T_STEPS 4096
#define BATCH   6
#define USE_ASYNC_PRELOAD 1

__device__ __forceinline__ u16 f2bf(float f) {
  union { float f; unsigned u; } x; x.f = f;
  unsigned u = x.u;
  u += 0x7fffu + ((u >> 16) & 1u);   // round to nearest even
  return (u16)(u >> 16);
}
__device__ __forceinline__ float bf2f(u16 b) {
  union { unsigned u; float f; } x; x.u = ((unsigned)b) << 16;
  return x.f;
}
__device__ __forceinline__ float sigf(float x) { return 1.0f / (1.0f + __expf(-x)); }

union FragU { uint4 q[2]; v16bf v; };

// A-fragment (16x32 bf16): lane l holds row m=l%16; 16 halves cover
// K = k0..k0+7 and k0+16..k0+23 with k0 = 8*(l/16)  (ISA 16-bit A layout).
__device__ __forceinline__ v16bf load_a_frag(const u16* rowbase, int k0) {
  FragU u;
  u.q[0] = *reinterpret_cast<const uint4*>(rowbase + k0);
  u.q[1] = *reinterpret_cast<const uint4*>(rowbase + k0 + 16);
  return u.v;
}

// --------------------------- prep kernels ----------------------------------

__global__ void zero_init_kernel(u16* zeros, unsigned* sync) {
  int i = threadIdx.x + blockIdx.x * blockDim.x;
  if (i < BATCH * 512) zeros[i] = 0;
  if (i < 8) sync[i] = 0;
}

__global__ void convert_x_kernel(const float* __restrict__ x, u16* __restrict__ xb, int n) {
  for (int i = threadIdx.x + blockIdx.x * blockDim.x; i < n; i += gridDim.x * blockDim.x)
    xb[i] = f2bf(x[i]);
}

// Pack [Wx;Wh] (K x 4H fp32, row-major) into per-(ntile,kblock) B-fragment
// blocks: blob[((ntile*KB + kb)*32 + lane)*16 + elem], lane -> n, elem -> k
// per the ISA 16-bit matrix layout (mirrors the A table with M->N).
__global__ void pack_weights_kernel(const float* __restrict__ Wx,
                                    const float* __restrict__ Wh,
                                    u16* __restrict__ blob, int Kx, int H) {
  const int fourH = 4 * H;
  const int K = Kx + H;
  const int KB = K / 32;
  const long long total = (long long)fourH * K;
  for (long long e = threadIdx.x + (long long)blockIdx.x * blockDim.x; e < total;
       e += (long long)gridDim.x * blockDim.x) {
    int elem = (int)(e & 15);
    int lane = (int)((e >> 4) & 31);
    long long frag = e >> 9;
    int kblock = (int)(frag % KB);
    int ntile  = (int)(frag / KB);
    int n = ntile * 16 + (lane & 15);
    int k = kblock * 32 + 8 * (lane >> 4) + elem + (elem >= 8 ? 8 : 0);
    float v = (k < Kx) ? Wx[(size_t)k * fourH + n] : Wh[(size_t)(k - Kx) * fourH + n];
    blob[e] = f2bf(v);
  }
}

// --------------------------- LSTM layer kernel -----------------------------

template <int KX, int HH, int USE_CINIT>
__global__ void __launch_bounds__(128) lstm_layer_kernel(
    const u16* __restrict__ in_seq,   // [T,6,KX] bf16
    const u16* __restrict__ h0,       // [6,HH]   bf16 (initial hidden)
    const u16* __restrict__ blob,     // packed weights, bf16
    const float* __restrict__ bias,   // [4*HH]   fp32 (gate order i,f,g,o)
    const float* __restrict__ c_init, // [6,HH]   fp32 (used if USE_CINIT)
    u16* __restrict__ out_seq,        // [T,6,HH] bf16 (doubles as h history)
    float* __restrict__ c_out,        // [6,HH]   fp32 (final cell state)
    unsigned* __restrict__ sync_cnt,  // grid barrier counter (pre-zeroed)
    int nwg) {
  constexpr int KTOT = KX + HH;
  constexpr int KB  = KTOT / 32;
  constexpr int KXB = KX / 32;
  constexpr int KHB = HH / 32;

  const int tid   = threadIdx.x;
  const int lane  = tid & 31;
  const int gate  = tid >> 5;        // wave 0..3 -> gates i,f,g,o
  const int lrow  = lane & 15;       // A row / B,D column within tile
  const int lhalf = lane >> 4;
  const int k0off = 8 * lhalf;
  const int unit  = blockIdx.x * 16 + lrow;       // hidden unit of this lane
  const int mrow  = (lrow < BATCH) ? lrow : 0;    // clamp padding rows -> row 0

  extern __shared__ u16 smem[];
  u16*   wlds = smem;                          // 4*KB*512 u16 weight fragments
  float* zlds = (float*)(smem + 4 * KB * 512); // 4*256 f32 gate exchange

  // ---- one-time: weights global -> LDS (resident for all T steps) ----
  {
    const int ntile = gate * (HH / 16) + blockIdx.x;
    const u16* src = blob + (size_t)ntile * KB * 512;
    u16* dst = wlds + (size_t)gate * KB * 512;
#if USE_ASYNC_PRELOAD
    for (int kb = 0; kb < KB; ++kb) {
      // the VDST VGPR gives the LDS address; generic-pointer low 32 bits are the
      // wave-relative LDS offset (flat aperture rule).
      unsigned ldsoff = (unsigned)(size_t)(dst + (kb * 32 + lane) * 16);
      unsigned long long ga = (unsigned long long)(size_t)(src + (kb * 32 + lane) * 16);
      asm volatile("global_load_async_to_lds_b128 %0, %1, off"
                   :: "v"(ldsoff), "v"(ga) : "memory");
      asm volatile("global_load_async_to_lds_b128 %0, %1, off"
                   :: "v"(ldsoff + 16u), "v"(ga + 16ull) : "memory");
    }
    asm volatile("s_wait_asynccnt 0x0" ::: "memory");
#else
    for (int kb = 0; kb < KB; ++kb) {
      const uint4* s = reinterpret_cast<const uint4*>(src + (kb * 32 + lane) * 16);
      uint4* d = reinterpret_cast<uint4*>(dst + (kb * 32 + lane) * 16);
      d[0] = s[0];
      d[1] = s[1];
    }
#endif
  }
  __syncthreads();

  const float bz = bias[gate * HH + blockIdx.x * 16 + lrow];
  const u16* wbase = wlds + (size_t)gate * KB * 512;
  const int  lslot = lane * 16 + k0off * 0;  // (compile-time folded below)

  // cell state tile (wave 0): element r, lane l <-> (m = r + 8*lhalf, unit)
  float cst[8];
#pragma unroll
  for (int r = 0; r < 8; ++r) {
    int m = r + 8 * lhalf;
    cst[r] = (USE_CINIT && m < BATCH) ? c_init[m * HH + unit] : 0.0f;
  }

  for (int t = 0; t < T_STEPS; ++t) {
    const u16* hsrc = (t == 0) ? h0 : (out_seq + (size_t)(t - 1) * BATCH * HH);
    const u16* xrow = in_seq + ((size_t)t * BATCH + mrow) * KX;
    const u16* hrow = hsrc + (size_t)mrow * HH;

    v8f acc;
#pragma unroll
    for (int r = 0; r < 8; ++r) acc[r] = bz;   // bias pre-added

    // x @ Wi slice : KXB blocks, branch-free
#pragma unroll
    for (int kb = 0; kb < KXB; ++kb) {
      v16bf a = load_a_frag(xrow, kb * 32 + k0off);
      FragU bu;
      const u16* wp = wbase + ((size_t)kb * 32 + lane) * 16;
      bu.q[0] = *reinterpret_cast<const uint4*>(wp);
      bu.q[1] = *reinterpret_cast<const uint4*>(wp + 8);
      acc = __builtin_amdgcn_wmma_f32_16x16x32_bf16(
          false, a, false, bu.v, (short)0, acc, false, false);
    }
    // h @ Wh slice : KHB blocks, branch-free
#pragma unroll
    for (int kb = 0; kb < KHB; ++kb) {
      v16bf a = load_a_frag(hrow, kb * 32 + k0off);
      FragU bu;
      const u16* wp = wbase + ((size_t)(KXB + kb) * 32 + lane) * 16;
      bu.q[0] = *reinterpret_cast<const uint4*>(wp);
      bu.q[1] = *reinterpret_cast<const uint4*>(wp + 8);
      acc = __builtin_amdgcn_wmma_f32_16x16x32_bf16(
          false, a, false, bu.v, (short)0, acc, false, false);
    }
    (void)lslot;

    // exchange gate tiles through LDS, lane-contiguous (2x ds_store_b128)
    *reinterpret_cast<v8f*>(zlds + gate * 256 + lane * 8) = acc;
    __syncthreads();

    if (gate == 0) {
      v8f zi = *reinterpret_cast<const v8f*>(zlds + 0 * 256 + lane * 8);
      v8f zf = *reinterpret_cast<const v8f*>(zlds + 1 * 256 + lane * 8);
      v8f zg = *reinterpret_cast<const v8f*>(zlds + 2 * 256 + lane * 8);
      v8f zo = *reinterpret_cast<const v8f*>(zlds + 3 * 256 + lane * 8);
#pragma unroll
      for (int r = 0; r < 8; ++r) {
        float ig = sigf(zi[r]), fg = sigf(zf[r]), gg = tanhf(zg[r]), og = sigf(zo[r]);
        cst[r] = fg * cst[r] + ig * gg;
        float h = og * tanhf(cst[r]);
        int m = r + 8 * lhalf;
        if (m < BATCH) out_seq[((size_t)t * BATCH + m) * HH + unit] = f2bf(h);
      }
    }

    // warm L2 for next step's activations (global_prefetch_b8)
    if (gate == 1 && lane < BATCH && (t + 1) < T_STEPS)
      __builtin_prefetch(in_seq + ((size_t)(t + 1) * BATCH + lane) * KX, 0, 1);

    __threadfence();
    __syncthreads();
    if (tid == 0) {
      __hip_atomic_fetch_add(sync_cnt, 1u, __ATOMIC_RELEASE, __HIP_MEMORY_SCOPE_AGENT);
      unsigned target = (unsigned)(t + 1) * (unsigned)nwg;
      while (__hip_atomic_load(sync_cnt, __ATOMIC_ACQUIRE, __HIP_MEMORY_SCOPE_AGENT) < target)
        __builtin_amdgcn_s_sleep(2);
    }
    __syncthreads();
  }

  // final cell state (layer chaining)
  if (gate == 0) {
#pragma unroll
    for (int r = 0; r < 8; ++r) {
      int m = r + 8 * lhalf;
      if (m < BATCH) c_out[m * HH + unit] = cst[r];
    }
  }
}

// --------------------------- final linear ----------------------------------

__global__ void final_linear_kernel(const u16* __restrict__ h_last, // [6,128] bf16
                                    const float* __restrict__ Wl,   // [128,1]
                                    const float* __restrict__ bl,   // [1]
                                    float* __restrict__ out) {
  int b = threadIdx.x;
  if (b < BATCH) {
    float s = bl[0];
    const u16* row = h_last + b * 128;
    for (int j = 0; j < 128; ++j) s += bf2f(row[j]) * Wl[j];
    out[b] = s;
  }
}

// --------------------------- host launch -----------------------------------

extern "C" void kernel_launch(void* const* d_in, const int* in_sizes, int n_in,
                              void* d_out, int out_size, void* d_ws, size_t ws_size,
                              hipStream_t stream) {
  (void)in_sizes; (void)n_in; (void)out_size; (void)ws_size;
  const float* x   = (const float*)d_in[0];
  const float* Wi1 = (const float*)d_in[1];
  const float* Wh1 = (const float*)d_in[2];
  const float* b1  = (const float*)d_in[3];
  const float* Wi2 = (const float*)d_in[4];
  const float* Wh2 = (const float*)d_in[5];
  const float* b2  = (const float*)d_in[6];
  const float* Wi3 = (const float*)d_in[7];
  const float* Wh3 = (const float*)d_in[8];
  const float* b3  = (const float*)d_in[9];
  const float* Wl  = (const float*)d_in[10];
  const float* bl  = (const float*)d_in[11];

  char* ws = (char*)d_ws;                       // ~67.2 MB used, all 256B-aligned
  u16*      xbf   = (u16*)(ws + 0);             //  6,291,456 B : x as bf16 [T,6,128]
  u16*      seq1  = (u16*)(ws + 6291456);       // 25,165,824 B : layer1 seq [T,6,512]
  u16*      seq2  = (u16*)(ws + 31457280);      // 25,165,824 B : layer2 seq
  u16*      seq3  = (u16*)(ws + 56623104);      //  6,291,456 B : layer3 seq [T,6,128]
  u16*      blob1 = (u16*)(ws + 62914560);      //  2,621,440 B : packed [Wi1;Wh1]
  u16*      blob2 = (u16*)(ws + 65536000);      //  4,194,304 B : packed [Wi2;Wh2]
  u16*      blob3 = (u16*)(ws + 69730304);      //    655,360 B : packed [Wi3;Wh3]
  u16*      zeros = (u16*)(ws + 70385664);      //      6,144 B : zero h0
  float*    cbuf  = (float*)(ws + 70391808);    //     12,288 B : chained c state
  unsigned* syncp = (unsigned*)(ws + 70404096); //        256 B : grid barriers

  zero_init_kernel<<<13, 256, 0, stream>>>(zeros, syncp);
  convert_x_kernel<<<1024, 256, 0, stream>>>(x, xbf, T_STEPS * BATCH * 128);
  pack_weights_kernel<<<2048, 256, 0, stream>>>(Wi1, Wh1, blob1, 128, 512);
  pack_weights_kernel<<<2048, 256, 0, stream>>>(Wi2, Wh2, blob2, 512, 512);
  pack_weights_kernel<<<1024, 256, 0, stream>>>(Wi3, Wh3, blob3, 512, 128);

  // layer 1: 32 WGs x 128 thr; LDS = 4*20 KB weights + 4 KB exchange
  lstm_layer_kernel<128, 512, 0><<<32, 128, 4 * 20 * 1024 + 4096, stream>>>(
      xbf, zeros, blob1, b1, cbuf, seq1, cbuf, syncp + 0, 32);
  // layer 2: inherits layer1 final (h = seq1[T-1], c = cbuf); LDS = 128KB + 4KB
  lstm_layer_kernel<512, 512, 1><<<32, 128, 4 * 32 * 1024 + 4096, stream>>>(
      seq1, seq1 + (size_t)(T_STEPS - 1) * BATCH * 512, blob2, b2, cbuf, seq2,
      cbuf, syncp + 1, 32);
  // layer 3: fresh state, H=128 -> 8 WGs
  lstm_layer_kernel<512, 128, 0><<<8, 128, 4 * 20 * 1024 + 4096, stream>>>(
      seq2, zeros, blob3, b3, cbuf, seq3, cbuf, syncp + 2, 8);

  final_linear_kernel<<<1, 64, 0, stream>>>(
      seq3 + (size_t)(T_STEPS - 1) * BATCH * 128, Wl, bl, (float*)d_out);
}
